// MultiHeadedSelfAttention_78632261255589
// MI455X (gfx1250) — matrix operand order repaired and checked by reading, not compiled, not run on hardware
//
#include <hip/hip_runtime.h>
#include <hip/hip_bf16.h>

// ---------------------------------------------------------------------------
// Types for CDNA5 WMMA
// ---------------------------------------------------------------------------
typedef __bf16 bf16_t;
typedef __attribute__((ext_vector_type(16))) __bf16 v16bf;
typedef __attribute__((ext_vector_type(8)))  __bf16 v8bf;
typedef __attribute__((ext_vector_type(8)))  float  v8f;
typedef __attribute__((ext_vector_type(4)))  unsigned int v4u;
typedef __attribute__((ext_vector_type(8)))  int v8i;
typedef __attribute__((ext_vector_type(4)))  int v4i;

#define WMMA_BF16(A, B, C) \
  __builtin_amdgcn_wmma_f32_16x16x32_bf16(false, (A), false, (B), (short)0, (C), false, false)

// Problem constants (match reference)
constexpr int Bn  = 2;
constexpr int Sn  = 2048;
constexpr int Dn  = 1024;
constexpr int Hn  = 16;
constexpr int HDn = 64;
constexpr int Rn  = 64;
constexpr int VOC = 2 * Rn + 1;   // 129
constexpr int Mtok = Bn * Sn;     // 4096
constexpr int VPADK = 144;        // vocab padded to 9 N-tiles for rbias GEMM
constexpr int VPADV = 160;        // vocab padded to 5 K-steps for w GEMM

// ---------------------------------------------------------------------------
// Fragment loaders (per CDNA5 ISA VGPR layouts, wave32)
// A-matrix 16-bit 16x32: lane = 16*lh + ll holds row M=ll,
//   K chunks {8*lh .. 8*lh+7} and {16+8*lh .. 16+8*lh+7}
// B-matrix 16-bit 32x16: lane holds col N=ll, K chunks {8*lh .. 8*lh+7} and {16+8*lh .. 16+8*lh+7}
// ---------------------------------------------------------------------------
__device__ __forceinline__ v16bf load_afrag(const bf16_t* p, int lh) {
  v8bf lo = *(const v8bf*)(p + 8 * lh);
  v8bf hi = *(const v8bf*)(p + 16 + 8 * lh);
  v16bf r;
#pragma unroll
  for (int i = 0; i < 8; ++i) { r[i] = lo[i]; r[i + 8] = hi[i]; }
  return r;
}
__device__ __forceinline__ v16bf load_bfrag(const bf16_t* p, int lh) {
  v8bf lo = *(const v8bf*)(p + 8 * lh);
  v8bf hi = *(const v8bf*)(p + 16 + 8 * lh);
  v16bf r;
#pragma unroll
  for (int i = 0; i < 8; ++i) { r[i] = lo[i]; r[i + 8] = hi[i]; }
  return r;
}

// ---------------------------------------------------------------------------
// Tensor Data Mover: 2D tile (rows x cols bf16) global -> LDS
// D# built per CDNA5 ISA (group0/group1 bitfields)
// ---------------------------------------------------------------------------
#if defined(__has_builtin)
#if __has_builtin(__builtin_amdgcn_tensor_load_to_lds)
#define HAVE_TDM 1
#endif
#endif

#ifdef HAVE_TDM
__device__ __forceinline__ void tdm_load_2d(unsigned lds_off, const bf16_t* gptr,
                                            int row_stride_elems, int rows_total,
                                            int tile_rows, int tile_cols) {
  unsigned long long ga = (unsigned long long)(uintptr_t)gptr;
  v4u g0;
  g0[0] = 1u;                                               // count=1, user desc
  g0[1] = lds_off;                                          // lds_addr [63:32]
  g0[2] = (unsigned)ga;                                     // global_addr lo
  g0[3] = ((unsigned)(ga >> 32) & 0x01FFFFFFu) | (2u << 30); // addr hi | type=2
  v8i g1;
  g1[0] = 0x00010000;                                       // data_size = 2B
  g1[1] = (int)(((unsigned)row_stride_elems & 0xFFFFu) << 16);           // tensor_dim0 lo
  g1[2] = (int)((((unsigned)row_stride_elems >> 16) & 0xFFFFu) |
                (((unsigned)rows_total & 0xFFFFu) << 16));               // td0 hi | td1 lo
  g1[3] = (int)((((unsigned)rows_total >> 16) & 0xFFFFu) |
                ((unsigned)tile_cols << 16));                            // td1 hi | tile_dim0
  g1[4] = tile_rows & 0xFFFF;                                            // tile_dim1 | tile_dim2=0
  g1[5] = row_stride_elems;                                              // tensor_dim0_stride lo32
  g1[6] = 0;
  g1[7] = 0;
  v4i z4 = {0, 0, 0, 0};
#if __clang_major__ >= 23
  v8i z8 = {0, 0, 0, 0, 0, 0, 0, 0};
  __builtin_amdgcn_tensor_load_to_lds(g0, g1, z4, z4, z8, 0);
#else
  __builtin_amdgcn_tensor_load_to_lds(g0, g1, z4, z4, 0);
#endif
}
#endif

// ---------------------------------------------------------------------------
// Prep kernels
// ---------------------------------------------------------------------------
__global__ void cvt_bf16_kernel(const float* __restrict__ src, bf16_t* __restrict__ dst, int n) {
  for (int i = blockIdx.x * blockDim.x + threadIdx.x; i < n; i += gridDim.x * blockDim.x)
    dst[i] = (bf16_t)src[i];
}

// src [K][N] fp32 row-major  ->  dst [N][K] bf16 (K contiguous = B-fragment layout)
__global__ void transpose_bf16_kernel(const float* __restrict__ src, bf16_t* __restrict__ dst,
                                      int K, int N) {
  for (int i = blockIdx.x * blockDim.x + threadIdx.x; i < K * N; i += gridDim.x * blockDim.x) {
    int k = i / N, n = i % N;
    dst[(size_t)n * K + k] = (bf16_t)src[i];
  }
}

__global__ void prep_rek_kernel(const float* __restrict__ rek, bf16_t* __restrict__ dst) {
  for (int i = blockIdx.x * blockDim.x + threadIdx.x; i < VPADK * HDn; i += gridDim.x * blockDim.x) {
    int v = i / HDn, h = i % HDn;
    dst[i] = (v < VOC) ? (bf16_t)rek[v * HDn + h] : (bf16_t)0.f;
  }
}

// rel_emb_v [VOC][HD] -> revT [HD][VPADV] (zero padded)
__global__ void prep_rev_kernel(const float* __restrict__ rev, bf16_t* __restrict__ dst) {
  for (int i = blockIdx.x * blockDim.x + threadIdx.x; i < HDn * VPADV; i += gridDim.x * blockDim.x) {
    int h = i / VPADV, m = i % VPADV;
    dst[i] = (m < VOC) ? (bf16_t)rev[m * HDn + h] : (bf16_t)0.f;
  }
}

// ---------------------------------------------------------------------------
// GEMM: C[M][N] = A[M][K](bf16) x Bt[N][K](bf16) + bias, fp32 accum via WMMA.
// Block: 256 threads = 8 waves; block tile 256x128; wave tile 64x64.
// A/B tiles staged to LDS by the Tensor Data Mover, double-buffered.
// MODE 0: QKV epilogue -> q(bf16, x0.125), k(bf16), v^T(bf16)
// MODE 1: out = val + bias (fp32)
// ---------------------------------------------------------------------------
template <int MODE>
__global__ __launch_bounds__(256) void gemm_bf16_wmma(
    const bf16_t* __restrict__ A, const bf16_t* __restrict__ Bt,
    const float* __restrict__ bias, int Ktot, int Ntot,
    bf16_t* __restrict__ qb, bf16_t* __restrict__ kb, bf16_t* __restrict__ vtb,
    float* __restrict__ outf) {
  extern __shared__ char smem[];
  bf16_t* Asm = (bf16_t*)smem;                      // [2][256][32]
  bf16_t* Bsm = (bf16_t*)(smem + 2 * 256 * 32 * 2); // [2][128][32]

  const int tid = threadIdx.x, wid = tid >> 5, lane = tid & 31;
  const int lh = lane >> 4, ll = lane & 15;
  const int m0 = blockIdx.x * 256, n0 = blockIdx.y * 128;
  const int wy = wid >> 1, wx = wid & 1;

  v8f acc[4][4] = {};
  const int nk = Ktot / 32;

#ifdef HAVE_TDM
  if (wid == 0) {
    tdm_load_2d(0u,     A  + (size_t)m0 * Ktot, Ktot, Mtok, 256, 32);
    tdm_load_2d(32768u, Bt + (size_t)n0 * Ktot, Ktot, Ntot, 128, 32);
  }
#endif

  for (int ks = 0; ks < nk; ++ks) {
    const int buf = ks & 1;
#ifdef HAVE_TDM
    if (wid == 0) __builtin_amdgcn_s_wait_tensorcnt(0);
    __syncthreads();
    if (wid == 0 && (ks + 1) < nk) {
      const int k1 = (ks + 1) * 32;
      tdm_load_2d((unsigned)((buf ^ 1) * 16384),
                  A + (size_t)m0 * Ktot + k1, Ktot, Mtok, 256, 32);
      tdm_load_2d((unsigned)(32768 + (buf ^ 1) * 8192),
                  Bt + (size_t)n0 * Ktot + k1, Ktot, Ntot, 128, 32);
    }
#else
    __syncthreads();
    {
      const int k0 = ks * 32;
      for (int i = tid; i < 256 * 32 / 8; i += 256) {
        int r = i >> 2, c = (i & 3) * 8;
        *(v8bf*)(Asm + buf * 8192 + r * 32 + c) =
            *(const v8bf*)(A + (size_t)(m0 + r) * Ktot + k0 + c);
      }
      for (int i = tid; i < 128 * 32 / 8; i += 256) {
        int r = i >> 2, c = (i & 3) * 8;
        *(v8bf*)(Bsm + buf * 4096 + r * 32 + c) =
            *(const v8bf*)(Bt + (size_t)(n0 + r) * Ktot + k0 + c);
      }
    }
    __syncthreads();
#endif
    const bf16_t* Ab = Asm + buf * 8192;
    const bf16_t* Bb = Bsm + buf * 4096;

    v16bf af[4], bfg[4];
#pragma unroll
    for (int t = 0; t < 4; ++t) af[t]  = load_afrag(Ab + (wy * 64 + t * 16 + ll) * 32, lh);
#pragma unroll
    for (int t = 0; t < 4; ++t) bfg[t] = load_bfrag(Bb + (wx * 64 + t * 16 + ll) * 32, lh);
#pragma unroll
    for (int i = 0; i < 4; ++i)
#pragma unroll
      for (int j = 0; j < 4; ++j)
        acc[i][j] = WMMA_BF16(af[i], bfg[j], acc[i][j]);
  }

  // Epilogue. C layout: lane holds col N=ll, rows M = r + 8*lh.
#pragma unroll
  for (int j = 0; j < 4; ++j) {
    const int n = n0 + wx * 64 + j * 16 + ll;
    const float bv = bias[n];
#pragma unroll
    for (int i = 0; i < 4; ++i) {
      const int mbase = m0 + wy * 64 + i * 16 + 8 * lh;
#pragma unroll
      for (int r = 0; r < 8; ++r) {
        const int t = mbase + r;
        const float val = acc[i][j][r] + bv;
        if (MODE == 0) {
          const int sel  = n >> 10;          // 0:q 1:k 2:v
          const int h    = (n & 1023) >> 6;
          const int hd   = n & 63;
          const int b    = t >> 11;
          const int srow = t & 2047;
          const int bh   = b * Hn + h;
          if (sel == 0)      qb [((size_t)bh * Sn + srow) * HDn + hd] = (bf16_t)(val * 0.125f);
          else if (sel == 1) kb [((size_t)bh * Sn + srow) * HDn + hd] = (bf16_t)val;
          else               vtb[((size_t)bh * HDn + hd) * Sn + srow] = (bf16_t)val;
        } else {
          outf[(size_t)t * Ntot + n] = val;
        }
      }
    }
  }
}

// ---------------------------------------------------------------------------
// Attention: one wave per 16 q-rows per (b,h). Block = 128 threads = 4 waves.
// S^T = K @ Q^T so softmax axis lies along registers; two-pass softmax
// (K/V are L2-resident on the 192MB L2, recompute of scores is cheap).
// Relative-position: rbias = qs @ rel_emb_k^T (gathered per score elem),
// value side via per-distance bucket sums: ctx += (bucketsum P) @ rel_emb_v.
// ---------------------------------------------------------------------------
__global__ __launch_bounds__(128) void attn_wmma(
    const bf16_t* __restrict__ qb, const bf16_t* __restrict__ kb,
    const bf16_t* __restrict__ vtb, const bf16_t* __restrict__ rekP,
    const bf16_t* __restrict__ revT, bf16_t* __restrict__ ctxb) {
  __shared__ __align__(64) bf16_t rb[4][16][VPADK];   // rbias[q][bucket]
  __shared__ __align__(64) float  wa[4][16][VPADV];   // bucket sums of probs
  __shared__ __align__(64) bf16_t pst[4][16][32];     // P staging (C->A relayout)
  __shared__ float ssum[4][16];

  const int wid = threadIdx.x >> 5, lane = threadIdx.x & 31;
  const int lh = lane >> 4, ll = lane & 15;
  const int bh = blockIdx.x;
  const int q0 = blockIdx.y * 64 + wid * 16;

  const bf16_t* Q  = qb  + (size_t)bh * Sn * HDn;
  const bf16_t* K  = kb  + (size_t)bh * Sn * HDn;
  const bf16_t* Vt = vtb + (size_t)bh * HDn * Sn;

  // Q fragments: A-layout (for rbias GEMM) and B-layout (as Q^T operand)
  v16bf aQ[2], bQ[2];
#pragma unroll
  for (int kk = 0; kk < 2; ++kk) {
    const bf16_t* qp = Q + (size_t)(q0 + ll) * HDn + kk * 32;
    aQ[kk] = load_afrag(qp, lh);
    bQ[kk] = load_bfrag(qp, lh);
  }

  // rbias = qs @ rel_emb_k^T  (16 x 144), 18 WMMAs
#pragma unroll
  for (int nt = 0; nt < 9; ++nt) {
    v8f c = {};
#pragma unroll
    for (int kk = 0; kk < 2; ++kk) {
      v16bf bb = load_bfrag(rekP + (size_t)(nt * 16 + ll) * HDn + kk * 32, lh);
      c = WMMA_BF16(aQ[kk], bb, c);
    }
#pragma unroll
    for (int r = 0; r < 8; ++r) rb[wid][r + 8 * lh][nt * 16 + ll] = (bf16_t)c[r];
  }

  // zero the bucket accumulator (wave-private region)
  for (int i = lane; i < 16 * VPADV; i += 32) ((float*)wa[wid])[i] = 0.f;
  asm volatile("s_wait_dscnt 0" ::: "memory");

  const int qg = q0 + ll;

  // ---- Pass A: running (max, sum) per lane; lane owns q=ll, key-half lh ----
  float mrun = -3.0e38f, srun = 0.f;
  for (int kt = 0; kt < Sn / 16; ++kt) {
    if (kt + 4 < Sn / 16) __builtin_prefetch(K + (size_t)(kt + 4) * 16 * HDn);
    v8f c = {};
#pragma unroll
    for (int kk = 0; kk < 2; ++kk) {
      v16bf ak = load_afrag(K + (size_t)(kt * 16 + ll) * HDn + kk * 32, lh);
      c = WMMA_BF16(ak, bQ[kk], c);   // S^T tile: (key = kt*16 + r + 8*lh, q = ll)
    }
    const int keyb = kt * 16 + 8 * lh;
    float sc[8];
    float tm = -3.0e38f;
#pragma unroll
    for (int r = 0; r < 8; ++r) {
      int dd = keyb + r - qg;
      dd = dd < -Rn ? -Rn : (dd > Rn ? Rn : dd);
      sc[r] = c[r] + (float)rb[wid][ll][dd + Rn];
      tm = fmaxf(tm, sc[r]);
    }
    const float nm = fmaxf(mrun, tm);
    float sl = 0.f;
#pragma unroll
    for (int r = 0; r < 8; ++r) sl += __expf(sc[r] - nm);
    srun = srun * __expf(mrun - nm) + sl;
    mrun = nm;
  }
  {  // combine the two key-halves of each q column (lanes l <-> l^16)
    const float m2 = __shfl_xor(mrun, 16, 32);
    const float s2 = __shfl_xor(srun, 16, 32);
    const float nm = fmaxf(mrun, m2);
    srun = srun * __expf(mrun - nm) + s2 * __expf(m2 - nm);
    mrun = nm;
  }
  if (lane < 16) ssum[wid][lane] = srun;
  asm volatile("s_wait_dscnt 0" ::: "memory");

  // ---- Pass B: probs, bucket sums, P @ V ----
  v8f acc[4] = {};
  for (int kb2 = 0; kb2 < Sn / 32; ++kb2) {
#pragma unroll
    for (int half = 0; half < 2; ++half) {
      const int kt = kb2 * 2 + half;
      v8f c = {};
#pragma unroll
      for (int kk = 0; kk < 2; ++kk) {
        v16bf ak = load_afrag(K + (size_t)(kt * 16 + ll) * HDn + kk * 32, lh);
        c = WMMA_BF16(ak, bQ[kk], c);
      }
      const int keyb = kt * 16 + 8 * lh;
      float p[8];
#pragma unroll
      for (int r = 0; r < 8; ++r) {
        int dd = keyb + r - qg;
        dd = dd < -Rn ? -Rn : (dd > Rn ? Rn : dd);
        const float scv = c[r] + (float)rb[wid][ll][dd + Rn];
        p[r] = __expf(scv - mrun);             // unnormalized prob
        atomicAdd(&wa[wid][ll][dd + Rn], p[r]); // bucket accumulate (ds_add_f32)
      }
#pragma unroll
      for (int r = 0; r < 8; ++r)
        pst[wid][ll][half * 16 + 8 * lh + r] = (bf16_t)p[r];
    }
    asm volatile("s_wait_dscnt 0" ::: "memory");
    v16bf ap = load_afrag(&pst[wid][ll][0], lh);   // P in A-layout (16q x 32key)
#pragma unroll
    for (int nt = 0; nt < 4; ++nt) {
      v16bf bv = load_bfrag(Vt + (size_t)(nt * 16 + ll) * Sn + kb2 * 32, lh);
      acc[nt] = WMMA_BF16(ap, bv, acc[nt]);
    }
  }

  // ---- ctx += w @ rel_emb_v (20 WMMAs) ----
  asm volatile("s_wait_dscnt 0" ::: "memory");
#pragma unroll
  for (int kk = 0; kk < VPADV / 32; ++kk) {
    v16bf aw;
#pragma unroll
    for (int i = 0; i < 8; ++i) {
      aw[i]     = (bf16_t)wa[wid][ll][kk * 32 + 8 * lh + i];
      aw[i + 8] = (bf16_t)wa[wid][ll][kk * 32 + 16 + 8 * lh + i];
    }
#pragma unroll
    for (int nt = 0; nt < 4; ++nt) {
      v16bf bb = load_bfrag(revT + (size_t)(nt * 16 + ll) * VPADV + kk * 32, lh);
      acc[nt] = WMMA_BF16(aw, bb, acc[nt]);
    }
  }

  // ---- normalize by row sums, store ctx (bf16, [B*S][D] row-major) ----
  float inv_s[8];
#pragma unroll
  for (int r = 0; r < 8; ++r) inv_s[r] = 1.f / ssum[wid][r + 8 * lh];
  const int b = bh >> 4, h = bh & 15;
#pragma unroll
  for (int nt = 0; nt < 4; ++nt) {
#pragma unroll
    for (int r = 0; r < 8; ++r) {
      const int srow = q0 + r + 8 * lh;
      ctxb[((size_t)(b * Sn + srow)) * Dn + h * HDn + nt * 16 + ll] =
          (bf16_t)(acc[nt][r] * inv_s[r]);
    }
  }
}

// ---------------------------------------------------------------------------
// Host launcher
// ---------------------------------------------------------------------------
extern "C" void kernel_launch(void* const* d_in, const int* in_sizes, int n_in,
                              void* d_out, int out_size, void* d_ws, size_t ws_size,
                              hipStream_t stream) {
  (void)in_sizes; (void)n_in; (void)out_size; (void)ws_size;
  const float* x    = (const float*)d_in[0];
  const float* Wqkv = (const float*)d_in[1];
  const float* bqkv = (const float*)d_in[2];
  const float* Wout = (const float*)d_in[3];
  const float* bout = (const float*)d_in[4];
  const float* rek  = (const float*)d_in[5];
  const float* rev  = (const float*)d_in[6];

  char* ws = (char*)d_ws;
  size_t off = 0;
  auto alloc = [&](size_t bytes) -> void* {
    void* p = ws + off;
    off += (bytes + 255) & ~(size_t)255;
    return p;
  };
  bf16_t* xb    = (bf16_t*)alloc((size_t)Mtok * Dn * 2);
  bf16_t* wqkvT = (bf16_t*)alloc((size_t)3 * Dn * Dn * 2);
  bf16_t* woutT = (bf16_t*)alloc((size_t)Dn * Dn * 2);
  bf16_t* qbuf  = (bf16_t*)alloc((size_t)Bn * Hn * Sn * HDn * 2);
  bf16_t* kbuf  = (bf16_t*)alloc((size_t)Bn * Hn * Sn * HDn * 2);
  bf16_t* vtbuf = (bf16_t*)alloc((size_t)Bn * Hn * Sn * HDn * 2);
  bf16_t* ctxb  = (bf16_t*)alloc((size_t)Mtok * Dn * 2);
  bf16_t* rekP  = (bf16_t*)alloc((size_t)VPADK * HDn * 2);
  bf16_t* revTb = (bf16_t*)alloc((size_t)HDn * VPADV * 2);

  cvt_bf16_kernel<<<2048, 256, 0, stream>>>(x, xb, Mtok * Dn);
  transpose_bf16_kernel<<<2048, 256, 0, stream>>>(Wqkv, wqkvT, Dn, 3 * Dn);
  transpose_bf16_kernel<<<2048, 256, 0, stream>>>(Wout, woutT, Dn, Dn);
  prep_rek_kernel<<<64, 256, 0, stream>>>(rek, rekP);
  prep_rev_kernel<<<64, 256, 0, stream>>>(rev, revTb);

  const size_t lds_bytes = 2 * 256 * 32 * 2 + 2 * 128 * 32 * 2;  // 48 KB double-buffered

  gemm_bf16_wmma<0><<<dim3(Mtok / 256, (3 * Dn) / 128), 256, lds_bytes, stream>>>(
      xb, wqkvT, bqkv, Dn, 3 * Dn, qbuf, kbuf, vtbuf, nullptr);

  attn_wmma<<<dim3(Bn * Hn, Sn / 64), 128, 0, stream>>>(qbuf, kbuf, vtbuf, rekP, revTb, ctxb);

  gemm_bf16_wmma<1><<<dim3(Mtok / 256, Dn / 128), 256, lds_bytes, stream>>>(
      ctxb, woutT, bout, Dn, Dn, nullptr, nullptr, nullptr, (float*)d_out);
}
